// SourceDetectLocalize_39032662786502
// MI455X (gfx1250) — matrix operand order repaired and checked by reading, not compiled, not run on hardware
//
#include <hip/hip_runtime.h>
#include <hip/hip_bf16.h>

typedef __attribute__((ext_vector_type(16))) __bf16 v16bf;
typedef __attribute__((ext_vector_type(8)))  __bf16 v8bf;
typedef __attribute__((ext_vector_type(4)))  __bf16 v4bf;
typedef __attribute__((ext_vector_type(8)))  float  v8f;

#define NBT    2048            // nb*nt rows (M)
#define FDIM   3072            // nf*nmic (K)
#define NG     16471           // nele*nazi (N)
#define NAZI_C 181
#define INV_SCALE (1.0f / 1536.0f)   // scale = nmic*nf/2 = 1536

// d_out float layout: DOAs (8,256,2,2) | VADs (8,256,2) | pred_ss (8,256,91,181)
#define DOA_OFF 0
#define VAD_OFF 8192
#define SS_OFF  12288

// ---------------------------------------------------------------- converts
__global__ void __launch_bounds__(256)
cvt_f32_to_bf16(const float* __restrict__ in, __bf16* __restrict__ out, long n4) {
    long i = (long)blockIdx.x * blockDim.x + threadIdx.x;
    if (i >= n4) return;
    float4 v = ((const float4*)in)[i];
    v4bf o;
    o[0] = (__bf16)v.x; o[1] = (__bf16)v.y;
    o[2] = (__bf16)v.z; o[3] = (__bf16)v.w;
    ((v4bf*)out)[i] = o;
}

__global__ void __launch_bounds__(256)
zero_u64(unsigned long long* __restrict__ p, int n) {
    int i = blockIdx.x * blockDim.x + threadIdx.x;
    if (i < n) p[i] = 0ull;
}

// ---------------------------------------------------------------- WMMA GEMM
// C[2048 x NG] = A[2048 x 3072] * B[NG x 3072]^T  (bf16 in, f32 acc)
// Block = 128 threads = 4 waves covering a 64M x 128N block tile.
// Wave w: M half = w&1 (32 rows), N half = w>>1 (64 cols).
// Per-wave tile = 32M x 64N = 2 A frags x 4 B frags -> 8 WMMA per K-step
// (12 b128 loads / 8 WMMA = 1.5 loads per WMMA).
__global__ void __launch_bounds__(128)
gemm_ss_argmax(const __bf16* __restrict__ A, const __bf16* __restrict__ B,
               float* __restrict__ ss, unsigned long long* __restrict__ amax,
               int write_ss)
{
    const int lane = threadIdx.x & 31;
    const int wave = threadIdx.x >> 5;
    const int r16  = lane & 15;
    const int half = lane >> 4;
    const int mbase = blockIdx.y * 64 + (wave & 1) * 32;
    const int nbase = blockIdx.x * 128 + (wave >> 1) * 64;

    // A fragment sources: lane r16 = row, half selects K sub-ranges
    // (ISA 7.12.2: 16-bit A 16x32 -> lane m holds K[h*8 .. +7] and K[16+h*8 ..])
    const __bf16* arow[2];
#pragma unroll
    for (int mi = 0; mi < 2; ++mi)
        arow[mi] = A + (size_t)(mbase + mi * 16 + r16) * FDIM;

    // B fragment sources: lane r16 = output column n (template row g), half
    // selects K[h*8 .. +7] and K[16+h*8 .. +7] (matches A, lanes = N cols)
    const __bf16* brow[4];
    int gcol[4];
#pragma unroll
    for (int t = 0; t < 4; ++t) {
        int g = nbase + t * 16 + r16;
        gcol[t] = g;
        brow[t] = B + (size_t)(g < NG ? g : (NG - 1)) * FDIM;
    }

    v8f acc[2][4] = {};
    for (int k0 = 0; k0 < FDIM; k0 += 32) {
        v16bf af[2];
#pragma unroll
        for (int mi = 0; mi < 2; ++mi) {
            v8bf a0 = *(const v8bf*)(arow[mi] + k0 + half * 8);
            v8bf a1 = *(const v8bf*)(arow[mi] + k0 + 16 + half * 8);
            af[mi] = __builtin_shufflevector(a0, a1,
                        0,1,2,3,4,5,6,7,8,9,10,11,12,13,14,15);
        }
#pragma unroll
        for (int t = 0; t < 4; ++t) {
            v8bf b0 = *(const v8bf*)(brow[t] + k0 + half * 8);
            v8bf b1 = *(const v8bf*)(brow[t] + k0 + 16 + half * 8);
            v16bf bf = __builtin_shufflevector(b0, b1,
                        0,1,2,3,4,5,6,7,8,9,10,11,12,13,14,15);
            acc[0][t] = __builtin_amdgcn_wmma_f32_16x16x32_bf16(
                            false, af[0], false, bf, (short)0, acc[0][t],
                            false, false);
            acc[1][t] = __builtin_amdgcn_wmma_f32_16x16x32_bf16(
                            false, af[1], false, bf, (short)0, acc[1][t],
                            false, false);
        }
    }

    // C layout (ISA 7.12.2): lane l, VGPR j -> row = j + 8*(l>=16), col = l&15
    // Non-temporal stores: keep the bf16 template L2-resident across passes.
    if (write_ss) {
#pragma unroll
        for (int mi = 0; mi < 2; ++mi) {
#pragma unroll
            for (int t = 0; t < 4; ++t) {
                if (gcol[t] < NG) {
#pragma unroll
                    for (int j = 0; j < 8; ++j) {
                        int r = mbase + mi * 16 + half * 8 + j;
                        __builtin_nontemporal_store(
                            acc[mi][t][j] * INV_SCALE,
                            ss + (size_t)r * NG + gcol[t]);
                    }
                }
            }
        }
    }

    // streaming per-row argmax over this wave's 64 columns, then atomic merge
#pragma unroll
    for (int mi = 0; mi < 2; ++mi) {
#pragma unroll
        for (int j = 0; j < 8; ++j) {
            float best = -__builtin_inff();
            int bidx = 0;
#pragma unroll
            for (int t = 0; t < 4; ++t) {
                if (gcol[t] < NG && acc[mi][t][j] > best) {
                    best = acc[mi][t][j]; bidx = gcol[t];
                }
            }
            // monotonic float key; ~idx so max() keeps the SMALLEST column on
            // value ties (jnp.argmax first-occurrence semantics)
            unsigned int b  = __float_as_uint(best);
            unsigned int hi = (b & 0x80000000u) ? ~b : (b | 0x80000000u);
            unsigned int lo = ~(unsigned int)bidx;
            // reduce across 16 lanes of each half (masks < 16 stay in-half)
#pragma unroll
            for (int m = 1; m < 16; m <<= 1) {
                unsigned int ohi = (unsigned int)__shfl_xor((int)hi, m, 32);
                unsigned int olo = (unsigned int)__shfl_xor((int)lo, m, 32);
                if (ohi > hi || (ohi == hi && olo > lo)) { hi = ohi; lo = olo; }
            }
            if (r16 == 0) {
                unsigned long long pk = (((unsigned long long)hi) << 32) | lo;
                atomicMax(amax + (mbase + mi * 16 + half * 8 + j), pk);
            }
        }
    }
}

// ---------------------------------------------------------------- finalize
// One 256-thread block per (b,t) row: decode argmax -> DOA, fp32 ratio/VAD,
// template subtraction into next-iteration ipd buffer.
__global__ void __launch_bounds__(256)
finalize_row(const unsigned long long* __restrict__ amax,
             const float* __restrict__ ipd_cur,
             const float* __restrict__ tpl,
             float* __restrict__ out,
             float* __restrict__ ipd_next,
             int kslot)
{
    __shared__ float s1[256];
    __shared__ float s2[256];
    __shared__ float sratio;

    const int row = blockIdx.x;
    const int tid = threadIdx.x;

    unsigned long long pk = amax[row];
    int g = (int)(~(unsigned int)(pk & 0xffffffffu));

    const float* tp = tpl + (size_t)g * FDIM;
    const float* ip = ipd_cur + (size_t)row * FDIM;

    float a = 0.0f, bb = 0.0f;
    for (int i = tid; i < FDIM; i += 256) {
        float tv = tp[i], iv = ip[i];
        a  += tv * iv;
        bb += tv * tv;
    }
    s1[tid] = a; s2[tid] = bb;
    __syncthreads();
    for (int s = 128; s > 0; s >>= 1) {
        if (tid < s) { s1[tid] += s1[tid + s]; s2[tid] += s2[tid + s]; }
        __syncthreads();
    }
    if (tid == 0) {
        float ratio = s1[0] / s2[0];
        sratio = ratio;
        out[VAD_OFF + row * 2 + kslot]           = ratio;
        out[DOA_OFF + (row * 2 + 0) * 2 + kslot] = (float)(g / NAZI_C) * 2.0f;
        out[DOA_OFF + (row * 2 + 1) * 2 + kslot] = (float)(g % NAZI_C) * 2.0f;
    }
    __syncthreads();
    if (ipd_next) {
        float ratio = sratio;
        float* op = ipd_next + (size_t)row * FDIM;
        for (int i = tid; i < FDIM; i += 256)
            op[i] = ip[i] - ratio * tp[i];
    }
}

// ---------------------------------------------------------------- launch
extern "C" void kernel_launch(void* const* d_in, const int* in_sizes, int n_in,
                              void* d_out, int out_size, void* d_ws, size_t ws_size,
                              hipStream_t stream) {
    (void)in_sizes; (void)n_in; (void)out_size; (void)ws_size;
    const float* pred_ipd = (const float*)d_in[0];
    const float* tpl      = (const float*)d_in[1];
    float* out = (float*)d_out;

    // workspace carve-up
    char* ws = (char*)d_ws;
    size_t off = 0;
    __bf16* tpl_bf = (__bf16*)(ws + off); off += (size_t)NG  * FDIM * 2; off = (off + 255) & ~(size_t)255;
    __bf16* ipd_bf = (__bf16*)(ws + off); off += (size_t)NBT * FDIM * 2; off = (off + 255) & ~(size_t)255;
    float*  ipd_f  = (float*) (ws + off); off += (size_t)NBT * FDIM * 4; off = (off + 255) & ~(size_t)255;
    unsigned long long* amax = (unsigned long long*)(ws + off);

    const long n4_tpl = ((long)NG  * FDIM) / 4;
    const long n4_ipd = ((long)NBT * FDIM) / 4;
    const int  blk_tpl = (int)((n4_tpl + 255) / 256);
    const int  blk_ipd = (int)((n4_ipd + 255) / 256);
    dim3 gemm_grid((NG + 127) / 128, NBT / 64);

    // ---- pass 0: pred_ss + argmax from original ipd
    zero_u64<<<(NBT + 255) / 256, 256, 0, stream>>>(amax, NBT);
    cvt_f32_to_bf16<<<blk_tpl, 256, 0, stream>>>(tpl,      tpl_bf, n4_tpl);
    cvt_f32_to_bf16<<<blk_ipd, 256, 0, stream>>>(pred_ipd, ipd_bf, n4_ipd);
    gemm_ss_argmax<<<gemm_grid, 128, 0, stream>>>(ipd_bf, tpl_bf,
                                                  out + SS_OFF, amax, 1);
    finalize_row<<<NBT, 256, 0, stream>>>(amax, pred_ipd, tpl, out, ipd_f, 0);

    // ---- pass 1: argmax on residual ipd
    zero_u64<<<(NBT + 255) / 256, 256, 0, stream>>>(amax, NBT);
    cvt_f32_to_bf16<<<blk_ipd, 256, 0, stream>>>(ipd_f, ipd_bf, n4_ipd);
    gemm_ss_argmax<<<gemm_grid, 128, 0, stream>>>(ipd_bf, tpl_bf,
                                                  out + SS_OFF, amax, 0);
    finalize_row<<<NBT, 256, 0, stream>>>(amax, ipd_f, tpl, out, nullptr, 1);
}
